// RNN_73186242723860
// MI455X (gfx1250) — compile-verified, matrix operand order checked by reading, not run on hardware
//
#include <hip/hip_runtime.h>

#define SEQ    128
#define BATCH  64
#define EMB    512
#define HID    1024
#define VOCAB  10000
#define VPAD   10240   // Wout rows padded so N tiles by 256 (10000 % 16 == 0, pad rows zeroed)

typedef __bf16 bf16_t;
typedef __attribute__((ext_vector_type(16))) __bf16 v16bf;
typedef __attribute__((ext_vector_type(8)))  __bf16 v8bf;
typedef __attribute__((ext_vector_type(8)))  float  v8f;
typedef __attribute__((ext_vector_type(4)))  unsigned int u32x4;
typedef __attribute__((ext_vector_type(8)))  int i32x8;
typedef __attribute__((ext_vector_type(4)))  int i32x4;

// Tensor Data Mover availability (device pass only; host pass falls back).
#if defined(__has_builtin)
#if __has_builtin(__builtin_amdgcn_tensor_load_to_lds) && \
    __has_builtin(__builtin_amdgcn_s_wait_tensorcnt)
#define USE_TDM 1
#endif
#endif
#ifndef USE_TDM
#define USE_TDM 0
#endif

// ---------------------------------------------------------------------------
// Load one 16x32 bf16 WMMA operand tile (A rows or B columns) from a
// row-major buffer whose K dimension is contiguous (stride = ld elements).
// ISA 7.12.2 16-bit layout: lanes 0-15 hold K {0..7, 16..23} of row (lane&15),
// lanes 16-31 hold K {8..15, 24..31}.  Two 16-byte loads per lane.
// ---------------------------------------------------------------------------
__device__ __forceinline__ v16bf load_frag16x32(const bf16_t* __restrict__ tile,
                                                int ld, int k0, int lane) {
  const int r  = lane & 15;
  const int ko = k0 + ((lane >> 4) << 3);
  const bf16_t* p = tile + (long long)r * ld + ko;
  v8bf lo = *(const v8bf*)(p);
  v8bf hi = *(const v8bf*)(p + 16);
  v16bf f;
#pragma unroll
  for (int i = 0; i < 8; ++i) { f[i] = lo[i]; f[i + 8] = hi[i]; }
  return f;
}

__device__ __forceinline__ v8f wmma_bf16(v16bf a, v16bf b, v8f c) {
  return __builtin_amdgcn_wmma_f32_16x16x32_bf16(false, a, false, b,
                                                 (short)0, c, false, false);
}

// ---------------------------------------------------------------------------
// Software-pipelined 16x64 GEMM segment: acc += X[m0:+16, :K] @ W[n0:+64, :K]^T
// X and W share row stride K (multiple of 64).  2x modulo-unrolled: two named
// register sets alternate, so no rotate copies and partial s_wait_loadcnt
// (next slice's loads stay in flight across the current WMMAs).
// ---------------------------------------------------------------------------
__device__ __forceinline__ void gemm_seg16x64(const bf16_t* __restrict__ X,
                                              const bf16_t* __restrict__ W,
                                              int K, int m0, int n0, int lane,
                                              v8f acc[4]) {
  const bf16_t* xrow = X + (long long)m0 * K;
  const bf16_t* wrow[4];
#pragma unroll
  for (int j = 0; j < 4; ++j) wrow[j] = W + (long long)(n0 + 16 * j) * K;

  v16bf a0 = load_frag16x32(xrow, K, 0, lane);
  v16bf b0[4];
#pragma unroll
  for (int j = 0; j < 4; ++j) b0[j] = load_frag16x32(wrow[j], K, 0, lane);

  for (int k0 = 0; k0 < K; k0 += 64) {
    // Phase 0: load set-1 @ k0+32 (always in range), compute with set-0.
    v16bf a1 = load_frag16x32(xrow, K, k0 + 32, lane);
    v16bf b1[4];
#pragma unroll
    for (int j = 0; j < 4; ++j) b1[j] = load_frag16x32(wrow[j], K, k0 + 32, lane);
#pragma unroll
    for (int j = 0; j < 4; ++j) acc[j] = wmma_bf16(a0, b0[j], acc[j]);

    // Phase 1: load set-0 @ k0+64 (guarded), compute with set-1.
    const int k2 = k0 + 64;
    if (k2 < K) {
      a0 = load_frag16x32(xrow, K, k2, lane);
#pragma unroll
      for (int j = 0; j < 4; ++j) b0[j] = load_frag16x32(wrow[j], K, k2, lane);
    }
#pragma unroll
    for (int j = 0; j < 4; ++j) acc[j] = wmma_bf16(a1, b1[j], acc[j]);
  }
}

// ---------------------------------------------------------------------------
// Elementwise f32 -> bf16 convert with row padding (pad rows -> 0).
// ---------------------------------------------------------------------------
__global__ void rnn_convert_pad(const float* __restrict__ src, bf16_t* __restrict__ dst,
                                int rows, int cols, int dstRows) {
  long long i = (long long)blockIdx.x * blockDim.x + threadIdx.x;
  long long total = (long long)dstRows * cols;
  if (i >= total) return;
  long long r = i / cols;
  dst[i] = (r < rows) ? (bf16_t)src[i] : (bf16_t)0.0f;
}

// ---------------------------------------------------------------------------
// Embedding gather: Xbf[s*B+b, :] = bf16(emb[idx[s,b], :])
// ---------------------------------------------------------------------------
__global__ void rnn_embed_gather(const int* __restrict__ idx, const float* __restrict__ emb,
                                 bf16_t* __restrict__ Xbf) {
  const int sb  = blockIdx.x;
  const int t   = threadIdx.x;
  const int row = idx[sb];
  const float4* src = (const float4*)(emb + (long long)row * EMB);
  bf16_t* dst = Xbf + (long long)sb * EMB;
  float4 v = src[t];
  dst[t * 4 + 0] = (bf16_t)v.x;
  dst[t * 4 + 1] = (bf16_t)v.y;
  dst[t * 4 + 2] = (bf16_t)v.z;
  dst[t * 4 + 3] = (bf16_t)v.w;
}

// ---------------------------------------------------------------------------
// Fused recurrence "diagonal": launch k computes
//   role 0 (blockIdx.z==0): h0_k    = tanh(x_k @ W0x^T + h0_{k-1} @ W0h^T + bh0)
//   role 1 (blockIdx.z==1): h1_{k-1}= tanh(h0_{k-1} @ W1x^T + h1_{k-2} @ W1h^T + bh1)
// Both roles only read h0_{k-1}; 129 launches cover 128 steps x 2 layers.
// One wave per 16(batch) x 64(feature) tile.  grid=(HID/64, BATCH/16, 2).
// ---------------------------------------------------------------------------
__global__ __launch_bounds__(32)
void rnn_step_fused(int k,
                    const bf16_t* __restrict__ Xbf,
                    const bf16_t* __restrict__ W0x, const bf16_t* __restrict__ W0h,
                    const float*  __restrict__ bh0,
                    const bf16_t* __restrict__ W1x, const bf16_t* __restrict__ W1h,
                    const float*  __restrict__ bh1,
                    bf16_t* __restrict__ h0pp,
                    const bf16_t* __restrict__ h1i,
                    bf16_t* __restrict__ H1,
                    float*  __restrict__ hidF) {
  const int role = blockIdx.z;
  if (role == 0 && k >= SEQ) return;   // last launch has no layer-0 work
  if (role == 1 && k == 0)   return;   // first launch has no layer-1 work

  const long long BH = (long long)BATCH * HID;
  const bf16_t *X, *Wx, *Wh, *Hp;
  const float* bias;
  bf16_t* Hout;
  float* HoutF = nullptr;
  int Kx;

  if (role == 0) {
    X    = Xbf + (long long)k * BATCH * EMB; Kx = EMB;
    Wx   = W0x; Wh = W0h; bias = bh0;
    Hp   = h0pp + (long long)((k + 1) & 1) * BH;   // h0_{k-1}
    Hout = h0pp + (long long)(k & 1) * BH;         // h0_k
    if (k == SEQ - 1) HoutF = hidF;                // final h0 (f32)
  } else {
    const int u = k - 1;
    X    = h0pp + (long long)(u & 1) * BH; Kx = HID;  // h0_u
    Wx   = W1x; Wh = W1h; bias = bh1;
    Hp   = (u == 0) ? h1i : (H1 + (long long)(u - 1) * BH);
    Hout = H1 + (long long)u * BH;
    if (u == SEQ - 1) HoutF = hidF + BH;           // final h1 (f32)
  }

  const int lane = threadIdx.x & 31;
  const int n0 = blockIdx.x * 64;
  const int m0 = blockIdx.y * 16;

  v8f acc[4];
#pragma unroll
  for (int j = 0; j < 4; ++j)
#pragma unroll
    for (int e = 0; e < 8; ++e) acc[j][e] = 0.0f;

  gemm_seg16x64(X,  Wx, Kx,  m0, n0, lane, acc);   // X  @ Wx^T
  gemm_seg16x64(Hp, Wh, HID, m0, n0, lane, acc);   // Hp @ Wh^T

  // Epilogue: bias + tanh.  C layout: VGPR e -> row e (lanes 0-15) / e+8.
  const int c  = lane & 15;
  const int r8 = (lane >> 4) << 3;
#pragma unroll
  for (int j = 0; j < 4; ++j) {
    const int col = n0 + 16 * j + c;
    const float bv = bias[col];
#pragma unroll
    for (int e = 0; e < 8; ++e) {
      const int row = m0 + e + r8;
      const float h = tanhf(acc[j][e] + bv);
      Hout[(long long)row * HID + col] = (bf16_t)h;
      if (HoutF) HoutF[(long long)row * HID + col] = h;
    }
  }
}

// ---------------------------------------------------------------------------
// TDM: issue a 2-D tensor_load_to_lds of a 128(row) x 32(K) bf16 slab of A
// into dynamic LDS at byte offset ldsOff.  Descriptor per ISA ch.8.
// ---------------------------------------------------------------------------
#if USE_TDM
__device__ __forceinline__ void tdm_load_slabA(const bf16_t* gptr, unsigned ldsOff) {
  unsigned long long ga = (unsigned long long)(uintptr_t)gptr;
  u32x4 g0;
  g0[0] = 1u;                                   // count=1, no gather
  g0[1] = ldsOff;                               // lds_addr (dynamic LDS base = 0)
  g0[2] = (unsigned)ga;                         // global_addr[31:0]
  g0[3] = (unsigned)(ga >> 32) | (2u << 30);    // global_addr[56:32] | type=2
  i32x8 g1;
  g1[0] = 0x00010000;                           // data_size=2B
  g1[1] = (int)(1024u << 16);                   // tensor_dim0 = 1024 (K extent)
  g1[2] = (int)(8192u << 16);                   // tensor_dim1 = 8192 rows
  g1[3] = (int)(32u << 16);                     // tile_dim0 = 32
  g1[4] = 128;                                  // tile_dim1 = 128, tile_dim2 = 0
  g1[5] = 1024;                                 // tensor_dim0_stride = HID
  g1[6] = 0;
  g1[7] = 0;
  i32x4 gz = {0, 0, 0, 0};                      // 2-D: groups 2/3 unused
#if __clang_major__ >= 23
  __builtin_amdgcn_tensor_load_to_lds(g0, g1, gz, gz, g1 * 0, 0);
#else
  __builtin_amdgcn_tensor_load_to_lds(g0, g1, gz, gz, 0);
#endif
}
#endif

// ---------------------------------------------------------------------------
// Output projection for all timesteps at once:
//   C[8192 x VOCAB] = H1[8192 x 1024] @ Wout^T + bout
// Workgroup tile 128(M) x 256(N); 8 waves as 2(M) x 4(N), each wave 64x64.
// K loop 2x-unrolled: slab ping-pong (TDM into the idle LDS buffer) and two
// named B-fragment register sets -> no rotate copies, partial waits.
// Logits streamed with non-temporal stores (327 MB; keep Wout/H1 in L2).
// grid=(VPAD/256, 8192/128), block=256, dynLDS = 16 KB.
// ---------------------------------------------------------------------------
#define SLAB_ELEMS (128 * 32)
#define SLAB_BYTES (SLAB_ELEMS * 2)

__global__ __launch_bounds__(256)
void rnn_logits_gemm(const bf16_t* __restrict__ A,    // [SEQ*BATCH x HID]
                     const bf16_t* __restrict__ B,    // [VPAD x HID]
                     const float*  __restrict__ bout, // [VOCAB]
                     float* __restrict__ C) {         // [SEQ*BATCH x VOCAB]
  extern __shared__ bf16_t As[];                      // 2 slabs, dyn-LDS base = 0

  const int tid  = threadIdx.x;
  const int lane = tid & 31;
  const int w    = tid >> 5;
  const int wm   = w & 1;
  const int wn   = w >> 1;
  const int m0wg = blockIdx.y * 128;
  const int m0   = wm * 64;
  const int n0   = blockIdx.x * 256 + wn * 64;

  const bf16_t* brow[4];
#pragma unroll
  for (int j = 0; j < 4; ++j) brow[j] = B + (long long)(n0 + 16 * j) * HID;
  const bf16_t* arow = A + (long long)m0wg * HID;

  v8f acc[4][4];
#pragma unroll
  for (int i = 0; i < 4; ++i)
#pragma unroll
    for (int j = 0; j < 4; ++j)
#pragma unroll
      for (int e = 0; e < 8; ++e) acc[i][j][e] = 0.0f;

#if !USE_TDM
  const int crow = tid >> 1;
  const int coff = (tid & 1) * 16;
#endif

  // Prime LDS buffer 0 with the k=0 A slab.
#if USE_TDM
  if (tid < 32) {
    tdm_load_slabA(arow, 0u);
    __builtin_amdgcn_s_wait_tensorcnt(0);
  }
#else
  {
    const uint4* g = (const uint4*)(arow + (long long)crow * HID + coff);
    uint4* s = (uint4*)(As + crow * 32 + coff);
    s[0] = g[0]; s[1] = g[1];
  }
#endif
  __syncthreads();

  // Prime B register set 0 for k=0.
  v16bf b0[4];
#pragma unroll
  for (int j = 0; j < 4; ++j) b0[j] = load_frag16x32(brow[j], HID, 0, lane);

  for (int k0 = 0; k0 < HID; k0 += 64) {
    // ---- Phase 0: compute k0 from slab 0 / set 0; prefill slab 1 & set 1 @ k0+32.
#if USE_TDM
    if (tid < 32) tdm_load_slabA(arow + (k0 + 32), SLAB_BYTES);
#else
    {
      const uint4* g = (const uint4*)(arow + (long long)crow * HID + (k0 + 32) + coff);
      uint4* s = (uint4*)(As + SLAB_ELEMS + crow * 32 + coff);
      s[0] = g[0]; s[1] = g[1];
    }
#endif
    v16bf b1[4];
#pragma unroll
    for (int j = 0; j < 4; ++j) b1[j] = load_frag16x32(brow[j], HID, k0 + 32, lane);

    v16bf a[4];
#pragma unroll
    for (int i = 0; i < 4; ++i)
      a[i] = load_frag16x32(As + (m0 + 16 * i) * 32, 32, 0, lane);
#pragma unroll
    for (int i = 0; i < 4; ++i)
#pragma unroll
      for (int j = 0; j < 4; ++j)
        acc[i][j] = wmma_bf16(a[i], b0[j], acc[i][j]);

#if USE_TDM
    if (tid < 32) __builtin_amdgcn_s_wait_tensorcnt(0);
#endif
    __syncthreads();

    // ---- Phase 1: compute k0+32 from slab 1 / set 1; prefill slab 0 & set 0 @ k0+64.
    const int k2 = k0 + 64;
    if (k2 < HID) {
#if USE_TDM
      if (tid < 32) tdm_load_slabA(arow + k2, 0u);
#else
      const uint4* g = (const uint4*)(arow + (long long)crow * HID + k2 + coff);
      uint4* s = (uint4*)(As + crow * 32 + coff);
      s[0] = g[0]; s[1] = g[1];
#endif
#pragma unroll
      for (int j = 0; j < 4; ++j) b0[j] = load_frag16x32(brow[j], HID, k2, lane);
      if (k2 + 32 < HID)
        __builtin_prefetch(brow[0] + k2 + 32, 0, 3);
    }

#pragma unroll
    for (int i = 0; i < 4; ++i)
      a[i] = load_frag16x32(As + SLAB_ELEMS + (m0 + 16 * i) * 32, 32, 0, lane);
#pragma unroll
    for (int i = 0; i < 4; ++i)
#pragma unroll
      for (int j = 0; j < 4; ++j)
        acc[i][j] = wmma_bf16(a[i], b1[j], acc[i][j]);

#if USE_TDM
    if (tid < 32) __builtin_amdgcn_s_wait_tensorcnt(0);
#endif
    __syncthreads();
  }

  const int c  = lane & 15;
  const int r8 = (lane >> 4) << 3;
#pragma unroll
  for (int i = 0; i < 4; ++i) {
#pragma unroll
    for (int j = 0; j < 4; ++j) {
      const int col = n0 + 16 * j + c;
      if (col < VOCAB) {
        const float bv = bout[col];
#pragma unroll
        for (int e = 0; e < 8; ++e) {
          const int row = m0wg + m0 + 16 * i + e + r8;
          __builtin_nontemporal_store(acc[i][j][e] + bv, &C[(long long)row * VOCAB + col]);
        }
      }
    }
  }
}

// ---------------------------------------------------------------------------
extern "C" void kernel_launch(void* const* d_in, const int* in_sizes, int n_in,
                              void* d_out, int out_size, void* d_ws, size_t ws_size,
                              hipStream_t stream) {
  const int*   inputs = (const int*)  d_in[0];   // [SEQ, BATCH]
  const float* hidden = (const float*)d_in[1];   // [2, BATCH, HID]
  const float* emb    = (const float*)d_in[2];   // [VOCAB, EMB]
  const float* Wih0   = (const float*)d_in[3];   // [HID, EMB]
  const float* Whh0   = (const float*)d_in[4];   // [HID, HID]
  const float* bh0    = (const float*)d_in[5];   // [HID]
  const float* Wih1   = (const float*)d_in[6];   // [HID, HID]
  const float* Whh1   = (const float*)d_in[7];   // [HID, HID]
  const float* bh1    = (const float*)d_in[8];   // [HID]
  const float* Wout   = (const float*)d_in[9];   // [VOCAB, HID]
  const float* bout   = (const float*)d_in[10];  // [VOCAB]
  float* out = (float*)d_out;                    // logits [S,B,V] ++ hidden_final [2,B,H]

  const long long BH = (long long)BATCH * HID;
  float* hid_final = out + (long long)SEQ * BATCH * VOCAB;

  char* ws = (char*)d_ws;
  auto carve = [&ws](long long elems) {
    bf16_t* p = (bf16_t*)ws;
    ws += elems * (long long)sizeof(bf16_t);
    return p;
  };
  bf16_t* Xbf  = carve((long long)SEQ * BATCH * EMB);
  bf16_t* W0x  = carve((long long)HID * EMB);
  bf16_t* W0h  = carve((long long)HID * HID);
  bf16_t* W1x  = carve((long long)HID * HID);
  bf16_t* W1h  = carve((long long)HID * HID);
  bf16_t* Wo   = carve((long long)VPAD * HID);
  bf16_t* h0pp = carve(2 * BH);
  bf16_t* h1i  = carve(BH);
  bf16_t* H1   = carve((long long)SEQ * BATCH * HID);

  auto cgrid = [](long long n) { return dim3((unsigned)((n + 255) / 256)); };

  rnn_convert_pad<<<cgrid((long long)HID * EMB),  256, 0, stream>>>(Wih0, W0x, HID, EMB, HID);
  rnn_convert_pad<<<cgrid((long long)HID * HID),  256, 0, stream>>>(Whh0, W0h, HID, HID, HID);
  rnn_convert_pad<<<cgrid((long long)HID * HID),  256, 0, stream>>>(Wih1, W1x, HID, HID, HID);
  rnn_convert_pad<<<cgrid((long long)HID * HID),  256, 0, stream>>>(Whh1, W1h, HID, HID, HID);
  rnn_convert_pad<<<cgrid((long long)VPAD * HID), 256, 0, stream>>>(Wout, Wo, VOCAB, HID, VPAD);
  rnn_convert_pad<<<cgrid(BH), 256, 0, stream>>>(hidden,      h0pp + BH, BATCH, HID, BATCH);
  rnn_convert_pad<<<cgrid(BH), 256, 0, stream>>>(hidden + BH, h1i,       BATCH, HID, BATCH);

  rnn_embed_gather<<<SEQ * BATCH, 128, 0, stream>>>(inputs, emb, Xbf);

  // Fused recurrence: 129 launches cover 128 steps x 2 layers.
  const dim3 sgrid(HID / 64, BATCH / 16, 2);
  for (int k = 0; k <= SEQ; ++k)
    rnn_step_fused<<<sgrid, 32, 0, stream>>>(k, Xbf, W0x, W0h, bh0,
                                             W1x, W1h, bh1, h0pp, h1i, H1, hid_final);

  // Hoisted output projection over all timesteps.
  rnn_logits_gemm<<<dim3(VPAD / 256, (SEQ * BATCH) / 128), 256,
                    2 * SLAB_BYTES, stream>>>(H1, Wo, bout, out);
}
